// LIVT_Decoder_74947179315523
// MI455X (gfx1250) — matrix operand order repaired and checked by reading, not compiled, not run on hardware
//
#include <hip/hip_runtime.h>
#include <hip/hip_bf16.h>
#include <math.h>

typedef __bf16 bf16_t;
typedef __attribute__((ext_vector_type(8)))  bf16_t bf16x8;
typedef __attribute__((ext_vector_type(16))) bf16_t v16bf;
typedef __attribute__((ext_vector_type(8)))  float  v8f;

#define HW    1024          // 32*32
#define CHW   6144          // 6*1024
#define QTOT  16384         // 128*128 queries
#define CHUNK 4096          // queries per chunk (A panel = 32.5 MB bf16)
#define KPAD  3968          // 3938 padded to multiple of 64
#define KIN   3938

__device__ __forceinline__ unsigned short f2bf(float f) {
    unsigned u = __float_as_uint(f);
    u += 0x7FFFu + ((u >> 16) & 1u);      // round-to-nearest-even
    return (unsigned short)(u >> 16);
}

// ---------------------------------------------------------------- conv3d 64->16
__global__ __launch_bounds__(256)
void conv_ch_kernel(const float* __restrict__ in, const float* __restrict__ W,
                    const float* __restrict__ b, float* __restrict__ out) {
    int t = blockIdx.x * blockDim.x + threadIdx.x;
    if (t >= 16 * CHW) return;
    int x = t & 31, y = (t >> 5) & 31, z = (t >> 10) % 6, oc = t / CHW;
    float acc = b[oc];
    for (int ic = 0; ic < 64; ++ic) {
        const float* ip = in + ic * CHW;
        const float* wp = W + (oc * 64 + ic) * 27;
#pragma unroll
        for (int kz = 0; kz < 3; ++kz) { int zz = z + kz - 1; if ((unsigned)zz >= 6u) continue;
#pragma unroll
            for (int ky = 0; ky < 3; ++ky) { int yy = y + ky - 1; if ((unsigned)yy >= 32u) continue;
#pragma unroll
                for (int kx = 0; kx < 3; ++kx) { int xx = x + kx - 1; if ((unsigned)xx >= 32u) continue;
                    acc = fmaf(ip[zz * HW + yy * 32 + xx], wp[kz * 9 + ky * 3 + kx], acc);
                }}}
    }
    out[t] = acc;
}

// ---------------------------------------------------------------- conv3d 16->16 (q,k,v fused)
__global__ __launch_bounds__(256)
void conv_qkv_kernel(const float* __restrict__ in,
                     const float* __restrict__ Wq, const float* __restrict__ bq,
                     const float* __restrict__ Wk, const float* __restrict__ bk,
                     const float* __restrict__ Wv, const float* __restrict__ bv,
                     float* __restrict__ outq, float* __restrict__ outk, float* __restrict__ outv) {
    int t = blockIdx.x * blockDim.x + threadIdx.x;
    if (t >= 16 * CHW) return;
    int x = t & 31, y = (t >> 5) & 31, z = (t >> 10) % 6, oc = t / CHW;
    float aq = bq[oc], ak = bk[oc], av = bv[oc];
    for (int ic = 0; ic < 16; ++ic) {
        const float* ip  = in + ic * CHW;
        const float* wpq = Wq + (oc * 16 + ic) * 27;
        const float* wpk = Wk + (oc * 16 + ic) * 27;
        const float* wpv = Wv + (oc * 16 + ic) * 27;
#pragma unroll
        for (int kz = 0; kz < 3; ++kz) { int zz = z + kz - 1; if ((unsigned)zz >= 6u) continue;
#pragma unroll
            for (int ky = 0; ky < 3; ++ky) { int yy = y + ky - 1; if ((unsigned)yy >= 32u) continue;
#pragma unroll
                for (int kx = 0; kx < 3; ++kx) { int xx = x + kx - 1; if ((unsigned)xx >= 32u) continue;
                    float v = ip[zz * HW + yy * 32 + xx];
                    int wi = kz * 9 + ky * 3 + kx;
                    aq = fmaf(v, wpq[wi], aq);
                    ak = fmaf(v, wpk[wi], ak);
                    av = fmaf(v, wpv[wi], av);
                }}}
    }
    outq[t] = aq; outk[t] = ak; outv[t] = av;
}

// ---------------------------------------------------------------- weight prep: K x N fp32 -> N x Kpad bf16 (transposed, zero-padded)
__global__ __launch_bounds__(256)
void prep_wt_kernel(const float* __restrict__ W, unsigned short* __restrict__ Wt,
                    int K, int N, int Kpad, int Npad) {
    int t = blockIdx.x * blockDim.x + threadIdx.x;
    if (t >= Kpad * Npad) return;
    int n = t / Kpad, k = t - n * Kpad;
    float v = (k < K && n < N) ? W[k * N + n] : 0.f;
    Wt[(size_t)n * Kpad + k] = f2bf(v);
}

// ---------------------------------------------------------------- attention: one thread per (query, dt, head)
__global__ __launch_bounds__(256)
void attn_kernel(const float* __restrict__ fq, const float* __restrict__ fk,
                 const float* __restrict__ fv, const float* __restrict__ times, int ti,
                 const float* __restrict__ Bpe, const float* __restrict__ Wpb,
                 const float* __restrict__ bpb, unsigned short* __restrict__ A, int p0) {
    int gid  = blockIdx.x * blockDim.x + threadIdx.x;
    int h    = gid & 7;
    int rest = gid >> 3;
    int dt   = rest % 5;
    int pl   = rest / 5;
    if (pl >= CHUNK) return;
    int p  = p0 + pl;
    int yq = p >> 7, xq = p & 127;
    float ycd = -1.f + (2.f * yq + 1.f) / 128.f;
    float xcd = -1.f + (2.f * xq + 1.f) / 128.f;
    float fyc = ((ycd + 1.f) * 32.f - 1.f) * 0.5f;
    float fxc = ((xcd + 1.f) * 32.f - 1.f) * 0.5f;
    int iy = (int)rintf(fyc); iy = iy < 0 ? 0 : (iy > 31 ? 31 : iy);
    int ix = (int)rintf(fxc); ix = ix < 0 ? 0 : (ix > 31 ? 31 : ix);

    float t5 = times[ti] * 5.f;
    int   ct = (int)rintf(t5);
    float rel_time = (t5 - (float)ct) * 0.4f;
    int frame = ct + dt - 2; frame = frame < 0 ? 0 : (frame > 5 ? 5 : frame);

    // trilinear sample of feat_q for channels 2h, 2h+1 (q-head)
    float iz  = ((rel_time + 1.f) * 5.f - 1.f) * 0.5f;
    float z0f = floorf(iz), y0f = floorf(fyc), x0f = floorf(fxc);
    float wz = iz - z0f, wy = fyc - y0f, wx = fxc - x0f;
    int z0 = (int)z0f, y0 = (int)y0f, x0 = (int)x0f;
    const float* q0p = fq + (2 * h) * CHW;
    const float* q1p = q0p + CHW;
    float sq0 = 0.f, sq1 = 0.f;
#pragma unroll
    for (int dz = 0; dz < 2; ++dz) {
        int zi = z0 + dz; float wzz = dz ? wz : (1.f - wz);
        bool vz = (zi >= 0) && (zi < 5);
        int zc = zi < 0 ? 0 : (zi > 4 ? 4 : zi);
        int fr = ct + zc - 2; fr = fr < 0 ? 0 : (fr > 5 ? 5 : fr);
#pragma unroll
        for (int dy = 0; dy < 2; ++dy) {
            int yi = y0 + dy; float wyy = dy ? wy : (1.f - wy);
            bool vy = (yi >= 0) && (yi < 32);
            int ycl = yi < 0 ? 0 : (yi > 31 ? 31 : yi);
#pragma unroll
            for (int dx = 0; dx < 2; ++dx) {
                int xi = x0 + dx; float wxx = dx ? wx : (1.f - wx);
                bool vx = (xi >= 0) && (xi < 32);
                int xcl = xi < 0 ? 0 : (xi > 31 ? 31 : xi);
                float w = wzz * wyy * wxx * ((vz && vy && vx) ? 1.f : 0.f);
                int off = fr * HW + ycl * 32 + xcl;
                sq0 = fmaf(w, q0p[off], sq0);
                sq1 = fmaf(w, q1p[off], sq1);
            }}}

    // separable positional-bias tables for this head
    float freq[10];
#pragma unroll
    for (int f = 0; f < 10; ++f) freq[f] = Bpe[f * 3];
    float tc = rel_time - (float)(dt - 2) * 0.4f;
    float At = bpb[h];
#pragma unroll
    for (int f = 0; f < 10; ++f) {
        float s, c; sincosf(tc * 5.f * freq[f], &s, &c);
        At = fmaf(s, Wpb[f * 8 + h], At);
        At = fmaf(c, Wpb[(30 + f) * 8 + h], At);
    }
    float cy = -1.f + (2.f * iy + 1.f) / 32.f;
    float cx = -1.f + (2.f * ix + 1.f) / 32.f;
    float ry32 = (ycd - cy) * 32.f;
    float rx32 = (xcd - cx) * 32.f;
    float Ay[7], Ax[7];
#pragma unroll
    for (int j = 0; j < 7; ++j) {
        float by = ry32 - 2.f * (float)(j - 3);
        float bx = rx32 - 2.f * (float)(j - 3);
        float ay = 0.f, ax = 0.f;
#pragma unroll
        for (int f = 0; f < 10; ++f) {
            float s, c;
            sincosf(by * freq[f], &s, &c);
            ay = fmaf(s, Wpb[(10 + f) * 8 + h], ay);
            ay = fmaf(c, Wpb[(40 + f) * 8 + h], ay);
            sincosf(bx * freq[f], &s, &c);
            ax = fmaf(s, Wpb[(20 + f) * 8 + h], ax);
            ax = fmaf(c, Wpb[(50 + f) * 8 + h], ax);
        }
        Ay[j] = ay; Ax[j] = ax;
    }

    const float* k0p = fk + (2 * h) * CHW + frame * HW;
    const float* k1p = k0p + CHW;
    const float* v0p = fv + (2 * h) * CHW + frame * HW;
    const float* v1p = v0p + CHW;
    float qh0 = sq0 * 0.70710678118654752f;
    float qh1 = sq1 * 0.70710678118654752f;

    // pass 1: max
    float m = -3.0e38f;
#pragma unroll
    for (int jy = 0; jy < 7; ++jy) {
        int yk = iy + jy - 3; bool vy = (yk >= 0) && (yk < 32);
#pragma unroll
        for (int jx = 0; jx < 7; ++jx) {
            int xk = ix + jx - 3; bool ok = vy && (xk >= 0) && (xk < 32);
            int off = (yk & 31) * 32 + (xk & 31);
            float kv0 = ok ? k0p[off] : 0.f;
            float kv1 = ok ? k1p[off] : 0.f;
            float sc = fmaf(qh0, kv0, fmaf(qh1, kv1, At + Ay[jy] + Ax[jx]));
            m = fmaxf(m, sc);
        }
    }
    // pass 2: sum of exp
    float ssum = 0.f;
#pragma unroll
    for (int jy = 0; jy < 7; ++jy) {
        int yk = iy + jy - 3; bool vy = (yk >= 0) && (yk < 32);
#pragma unroll
        for (int jx = 0; jx < 7; ++jx) {
            int xk = ix + jx - 3; bool ok = vy && (xk >= 0) && (xk < 32);
            int off = (yk & 31) * 32 + (xk & 31);
            float kv0 = ok ? k0p[off] : 0.f;
            float kv1 = ok ? k1p[off] : 0.f;
            float sc = fmaf(qh0, kv0, fmaf(qh1, kv1, At + Ay[jy] + Ax[jx]));
            ssum += __expf(sc - m);
        }
    }
    float inv = 1.f / ssum;
    // pass 3: write attn-weighted V into activation matrix (bf16)
    unsigned short* Ab = A + (size_t)pl * KPAD + dt * 784 + 2 * h;
#pragma unroll
    for (int jy = 0; jy < 7; ++jy) {
        int yk = iy + jy - 3; bool vy = (yk >= 0) && (yk < 32);
#pragma unroll
        for (int jx = 0; jx < 7; ++jx) {
            int xk = ix + jx - 3; bool ok = vy && (xk >= 0) && (xk < 32);
            int off = (yk & 31) * 32 + (xk & 31);
            float kv0 = ok ? k0p[off] : 0.f;
            float kv1 = ok ? k1p[off] : 0.f;
            float sc = fmaf(qh0, kv0, fmaf(qh1, kv1, At + Ay[jy] + Ax[jx]));
            float a  = __expf(sc - m) * inv;
            float vv0 = ok ? v0p[off] : 0.f;
            float vv1 = ok ? v1p[off] : 0.f;
            int k = jy * 7 + jx;
            Ab[k * 16]     = f2bf(vv0 * a);
            Ab[k * 16 + 1] = f2bf(vv1 * a);
        }
    }
    // tail columns: sq (16), rel_cell (2), zero pad (30)
    if (dt == 0) {
        unsigned short* T = A + (size_t)pl * KPAD + 3920;
        T[2 * h]     = f2bf(sq0);
        T[2 * h + 1] = f2bf(sq1);
        if (h == 0) {
            T[16] = 0x3F00; T[17] = 0x3F00;   // 0.5f in bf16
#pragma unroll
            for (int j = 18; j < 48; ++j) T[j] = 0;
        }
    }
}

// ---------------------------------------------------------------- WMMA fragment loads
__device__ __forceinline__ v16bf load_a_frag(const bf16_t* p) {
    // 16-bit A layout (matches B): contiguous 16-run of K at 16*kh per row lane
    bf16x8 lo = *reinterpret_cast<const bf16x8*>(p);
    bf16x8 hi = *reinterpret_cast<const bf16x8*>(p + 8);
    v16bf a;
#pragma unroll
    for (int i = 0; i < 8; ++i) { a[i] = lo[i]; a[i + 8] = hi[i]; }
    return a;
}
__device__ __forceinline__ v16bf load_b_frag(const bf16_t* p) {
    // 16-bit B layout used here (matches A): contiguous 16-run of K at 16*kh per column lane
    bf16x8 b0 = *reinterpret_cast<const bf16x8*>(p);
    bf16x8 b1 = *reinterpret_cast<const bf16x8*>(p + 8);
    v16bf b;
#pragma unroll
    for (int i = 0; i < 8; ++i) { b[i] = b0[i]; b[i + 8] = b1[i]; }
    return b;
}

// ---------------------------------------------------------------- WMMA GEMM: out = act(A[MxK] @ Wt^T + bias)
// A: M x K bf16 row-major; Wt: N x K bf16 (pre-transposed).
// One wave = (16*MT) x (16*NACC) tile. K-loop phase-unrolled by 64 (ping-pong
// fragment sets -> no rotation moves, loads of one phase overlap WMMAs of the other).
// Requires K % 64 == 0 (3968 = 62*64, 256 = 4*64).
template <int MT, int NACC, bool GELU>
__global__ __launch_bounds__(256)
void gemm_wmma_kernel(const unsigned short* __restrict__ Abf,
                      const unsigned short* __restrict__ Wt,
                      const float* __restrict__ bias,
                      int M, int N, int K,
                      unsigned short* __restrict__ outbf,
                      float* __restrict__ outf, int p0) {
    int gwave = (int)((blockIdx.x * blockDim.x + threadIdx.x) >> 5);
    int lane  = threadIdx.x & 31;
    int ngroups = N / (16 * NACC);
    int tm = gwave / ngroups, tg = gwave - tm * ngroups;
    int m0 = tm * (16 * MT), n0 = tg * (16 * NACC);
    if (m0 >= M) return;
    int l15 = lane & 15, kh = lane >> 4;
    const bf16_t* Ap[MT];
#pragma unroll
    for (int i = 0; i < MT; ++i)
        Ap[i] = reinterpret_cast<const bf16_t*>(Abf) + (size_t)(m0 + 16 * i + l15) * K + (kh << 4);
    const bf16_t* Bp[NACC];
#pragma unroll
    for (int j = 0; j < NACC; ++j)
        Bp[j] = reinterpret_cast<const bf16_t*>(Wt) + (size_t)(n0 + 16 * j + l15) * K + (kh << 4);

    v8f acc[MT][NACC] = {};
    v16bf a_f[2][MT], b_f[2][NACC];
    // prologue: phase 0 fragments at k=0
#pragma unroll
    for (int i = 0; i < MT; ++i) a_f[0][i] = load_a_frag(Ap[i]);
#pragma unroll
    for (int j = 0; j < NACC; ++j) b_f[0][j] = load_b_frag(Bp[j]);

    for (int k0 = 0; k0 < K; k0 += 64) {
        int k1 = k0 + 32;                              // always < K (K % 64 == 0)
        int k2 = (k0 + 64 < K) ? (k0 + 64) : 0;        // wrap keeps addresses valid
        __builtin_prefetch(Ap[0] + k0 + 512, 0, 1);
        // load phase 1 while computing phase 0
#pragma unroll
        for (int i = 0; i < MT; ++i) a_f[1][i] = load_a_frag(Ap[i] + k1);
#pragma unroll
        for (int j = 0; j < NACC; ++j) b_f[1][j] = load_b_frag(Bp[j] + k1);
#pragma unroll
        for (int i = 0; i < MT; ++i)
#pragma unroll
            for (int j = 0; j < NACC; ++j)
                acc[i][j] = __builtin_amdgcn_wmma_f32_16x16x32_bf16(
                    false, a_f[0][i], false, b_f[0][j], (short)0, acc[i][j], false, false);
        // load phase 0 (next outer iter) while computing phase 1
#pragma unroll
        for (int i = 0; i < MT; ++i) a_f[0][i] = load_a_frag(Ap[i] + k2);
#pragma unroll
        for (int j = 0; j < NACC; ++j) b_f[0][j] = load_b_frag(Bp[j] + k2);
#pragma unroll
        for (int i = 0; i < MT; ++i)
#pragma unroll
            for (int j = 0; j < NACC; ++j)
                acc[i][j] = __builtin_amdgcn_wmma_f32_16x16x32_bf16(
                    false, a_f[1][i], false, b_f[1][j], (short)0, acc[i][j], false, false);
    }

#pragma unroll
    for (int i = 0; i < MT; ++i) {
        int rowb = m0 + 16 * i + (kh << 3);   // C layout: VGPR v -> M = v + 8*(lane>>4)
#pragma unroll
        for (int j = 0; j < NACC; ++j) {
            int col = n0 + j * 16 + l15;
            float bc = GELU ? bias[col] : (col < 3 ? bias[col] : 0.f);
#pragma unroll
            for (int v = 0; v < 8; ++v) {
                float x = acc[i][j][v] + bc;
                int r = rowb + v;
                if (GELU) {
                    x = 0.5f * x * (1.f + erff(x * 0.70710678118654752f));
                    outbf[(size_t)r * N + col] = f2bf(x);
                } else if (col < 3) {
                    outf[(size_t)col * QTOT + p0 + r] = x;   // (3,128,128) transpose
                }
            }
        }
    }
}

// ---------------------------------------------------------------- host orchestration
extern "C" void kernel_launch(void* const* d_in, const int* in_sizes, int n_in,
                              void* d_out, int out_size, void* d_ws, size_t ws_size,
                              hipStream_t stream) {
    const float* feat_img = (const float*)d_in[0];
    const float* times    = (const float*)d_in[2];
    const float* W_ch = (const float*)d_in[3];  const float* b_ch = (const float*)d_in[4];
    const float* W_q  = (const float*)d_in[5];  const float* b_q  = (const float*)d_in[6];
    const float* W_k  = (const float*)d_in[7];  const float* b_k  = (const float*)d_in[8];
    const float* W_v  = (const float*)d_in[9];  const float* b_v  = (const float*)d_in[10];
    const float* B_pe = (const float*)d_in[11];
    const float* W_pb = (const float*)d_in[12]; const float* b_pb = (const float*)d_in[13];
    const float* W1 = (const float*)d_in[14]; const float* b1 = (const float*)d_in[15];
    const float* W2 = (const float*)d_in[16]; const float* b2 = (const float*)d_in[17];
    const float* W3 = (const float*)d_in[18]; const float* b3 = (const float*)d_in[19];
    const float* W4 = (const float*)d_in[20]; const float* b4 = (const float*)d_in[21];
    const float* W5 = (const float*)d_in[22]; const float* b5 = (const float*)d_in[23];

    char* ws = (char*)d_ws;
    float* feat  = (float*)(ws + 0);
    float* featq = (float*)(ws + 393216);
    float* featk = (float*)(ws + 786432);
    float* featv = (float*)(ws + 1179648);
    unsigned short* Wt1  = (unsigned short*)(ws + 1572864);   // 256 x 3968 bf16
    unsigned short* Wt2  = (unsigned short*)(ws + 3604480);
    unsigned short* Wt3  = (unsigned short*)(ws + 3735552);
    unsigned short* Wt4  = (unsigned short*)(ws + 3866624);
    unsigned short* Wt5  = (unsigned short*)(ws + 3997696);   // 16 x 256 bf16
    unsigned short* Abuf = (unsigned short*)(ws + 4005888);   // 4096 x 3968 bf16
    unsigned short* H1   = (unsigned short*)(ws + 36511744);  // 4096 x 256 bf16
    unsigned short* H2   = (unsigned short*)(ws + 38608896);

    conv_ch_kernel<<<384, 256, 0, stream>>>(feat_img, W_ch, b_ch, feat);
    conv_qkv_kernel<<<384, 256, 0, stream>>>(feat, W_q, b_q, W_k, b_k, W_v, b_v,
                                             featq, featk, featv);
    prep_wt_kernel<<<3968, 256, 0, stream>>>(W1, Wt1, KIN, 256, KPAD, 256);
    prep_wt_kernel<<<256, 256, 0, stream>>>(W2, Wt2, 256, 256, 256, 256);
    prep_wt_kernel<<<256, 256, 0, stream>>>(W3, Wt3, 256, 256, 256, 256);
    prep_wt_kernel<<<256, 256, 0, stream>>>(W4, Wt4, 256, 256, 256, 256);
    prep_wt_kernel<<<16, 256, 0, stream>>>(W5, Wt5, 256, 3, 256, 16);

    int n_times = in_sizes[2];
    for (int ti = 0; ti < n_times; ++ti) {
        float* outp = (float*)d_out + (size_t)ti * 3 * QTOT;
        for (int cch = 0; cch < QTOT / CHUNK; ++cch) {
            int p0 = cch * CHUNK;
            attn_kernel<<<640, 256, 0, stream>>>(featq, featk, featv, times, ti,
                                                 B_pe, W_pb, b_pb, Abuf, p0);
            // 8 waves/block; wave tile 32x64: (4096/32)*(256/64)=512 waves -> 64 blocks
            gemm_wmma_kernel<2, 4, true ><<<64, 256, 0, stream>>>(Abuf, Wt1, b1, CHUNK, 256, KPAD, H1, nullptr, 0);
            gemm_wmma_kernel<2, 4, true ><<<64, 256, 0, stream>>>(H1,   Wt2, b2, CHUNK, 256, 256,  H2, nullptr, 0);
            gemm_wmma_kernel<2, 4, true ><<<64, 256, 0, stream>>>(H2,   Wt3, b3, CHUNK, 256, 256,  H1, nullptr, 0);
            gemm_wmma_kernel<2, 4, true ><<<64, 256, 0, stream>>>(H1,   Wt4, b4, CHUNK, 256, 256,  H2, nullptr, 0);
            // final: wave tile 32x16: (4096/32)*1 = 128 waves -> 16 blocks
            gemm_wmma_kernel<2, 1, false><<<16, 256, 0, stream>>>(H2,   Wt5, b5, CHUNK, 16,  256,  nullptr, outp, p0);
        }
    }
}
